// GAT_40166534152610
// MI455X (gfx1250) — matrix operand order repaired and checked by reading, not compiled, not run on hardware
//
#include <hip/hip_runtime.h>
#include <hip/hip_bf16.h>
#include <cstdint>

typedef __attribute__((ext_vector_type(16))) __bf16 v16bf;
typedef __attribute__((ext_vector_type(8)))  __bf16 v8bf;
typedef __attribute__((ext_vector_type(8)))  float  v8f;

#define LRELU_ALPHA 0.2f
#define NB   4096          // nodes
#define DD   64            // head dim
#define HH   8
#define BB   2
#define ROWS_PER_WG 128    // 8 waves x 16 rows
#define TM   64            // m-tile (2 WMMA K-steps)
#define WPITCH 72          // bf16 elems per LDS row = 144B (16B aligned, conflict-friendly)

__device__ __forceinline__ unsigned f32_mono(float f) {
  unsigned u = __float_as_uint(f);
  return (u & 0x80000000u) ? ~u : (u | 0x80000000u);
}
__device__ __forceinline__ float mono_f32(unsigned u) {
  unsigned b = (u & 0x80000000u) ? (u & 0x7fffffffu) : ~u;
  return __uint_as_float(b);
}

__global__ void k_init(unsigned* __restrict__ m2u) {
  int t = threadIdx.x;
  if (t < BB * HH) m2u[t] = 0u;   // 0 < mono(x) for every real x
}

// ---------------------------------------------------------------------------
// Pass 1: Wh = hh @ W  (bf16 WMMA), emit Wh^T (bf16), Wh1/Wh2 (f32), max(Wh2)
// grid = (B*H) * (N/128), block = 256 (8 waves)
// ---------------------------------------------------------------------------
__global__ __launch_bounds__(256) void k_pass1(
    const float* __restrict__ h, const float* __restrict__ W,
    const float* __restrict__ a,
    __bf16* __restrict__ whT, float* __restrict__ wh1, float* __restrict__ wh2,
    unsigned* __restrict__ m2u)
{
  __shared__ __align__(16) __bf16 wtl[DD * WPITCH];   // W[hd]^T, [o][k] bf16
  __shared__ float whl[ROWS_PER_WG * 65];             // Wh tile f32, pitch 65

  const int bx  = blockIdx.x;
  const int bh  = bx >> 5;
  const int nb  = bx & 31;
  const int b   = bh >> 3;
  const int hd  = bh & 7;
  const int n0  = nb * ROWS_PER_WG;
  const int t   = threadIdx.x;
  const int lane = t & 31;
  const int w   = t >> 5;

  // stage W[hd] transposed into LDS as bf16: wtl[o*WPITCH + k] = W[hd][k][o]
  {
    const float* wp = W + hd * DD * DD;
    const int k  = t >> 2;          // 0..63
    const int o0 = (t & 3) * 16;
#pragma unroll
    for (int i = 0; i < 16; ++i)
      wtl[(o0 + i) * WPITCH + k] = (__bf16)wp[k * DD + o0 + i];
  }
  __syncthreads();

  const int half   = lane >> 4;     // 0/1
  const int lrow16 = lane & 15;
  const int nw0    = n0 + w * 16;
  const int nrow   = nw0 + lrow16;

  v8f acc[4] = {};
  const float* hp = h + ((size_t)(b * NB + nrow)) * (HH * DD) + hd * DD;

#pragma unroll
  for (int s = 0; s < 2; ++s) {
    // A operand: 16x32 bf16, lane row = lrow16, K = half*8 + g*16 + j
    v16bf av;
#pragma unroll
    for (int g = 0; g < 2; ++g) {
      const float4 f0 = *(const float4*)(hp + s * 32 + half * 8 + g * 16);
      const float4 f1 = *(const float4*)(hp + s * 32 + half * 8 + g * 16 + 4);
      av[g * 8 + 0] = (__bf16)f0.x; av[g * 8 + 1] = (__bf16)f0.y;
      av[g * 8 + 2] = (__bf16)f0.z; av[g * 8 + 3] = (__bf16)f0.w;
      av[g * 8 + 4] = (__bf16)f1.x; av[g * 8 + 5] = (__bf16)f1.y;
      av[g * 8 + 6] = (__bf16)f1.z; av[g * 8 + 7] = (__bf16)f1.w;
    }
#pragma unroll
    for (int ct = 0; ct < 4; ++ct) {
      const int o = ct * 16 + lrow16;            // B column
      const __bf16* bp = &wtl[o * WPITCH + s * 32 + half * 8];
      v8bf blo = *(const v8bf*)(bp);
      v8bf bhi = *(const v8bf*)(bp + 16);
      v16bf bv = __builtin_shufflevector(blo, bhi,
                   0,1,2,3,4,5,6,7,8,9,10,11,12,13,14,15);
      acc[ct] = __builtin_amdgcn_wmma_f32_16x16x32_bf16(
                  false, av, false, bv, (short)0, acc[ct], false, false);
    }
  }

  // D tile: lane holds (M = half*8+r, o = ct*16+lrow16) -> Wh^T rows directly
  __bf16* wtg = whT + (size_t)bh * DD * NB;
#pragma unroll
  for (int ct = 0; ct < 4; ++ct) {
    v8bf vb;
#pragma unroll
    for (int r = 0; r < 8; ++r) {
      const int M = half * 8 + r;
      whl[(w * 16 + M) * 65 + ct * 16 + lrow16] = acc[ct][r];
      vb[r] = (__bf16)acc[ct][r];
    }
    const int o = ct * 16 + lrow16;
    *(v8bf*)(wtg + (size_t)o * NB + nw0 + half * 8) = vb;   // 16B store
  }
  __syncthreads();

  if (t < ROWS_PER_WG) {
    const float* row = &whl[t * 65];
    const float* ap  = a + hd * 2 * DD;
    float s1 = 0.f, s2 = 0.f;
#pragma unroll 8
    for (int j = 0; j < DD; ++j) {
      const float v = row[j];
      s1 += v * ap[j];
      s2 += v * ap[DD + j];
    }
    const int gi = bh * NB + n0 + t;
    wh1[gi] = s1;
    wh2[gi] = s2;
    atomicMax(&m2u[bh], f32_mono(s2));   // deterministic (idempotent max)
  }
}

// ---------------------------------------------------------------------------
// Pass 2: flash-attention with exact precomputed row-max (lrelu monotone)
// grid = (B*H) * (N/128), block = 256 (8 waves)
// ---------------------------------------------------------------------------
__global__ __launch_bounds__(256) void k_pass2(
    const __bf16* __restrict__ whT, const float* __restrict__ wh1,
    const float* __restrict__ wh2, const unsigned* __restrict__ m2u,
    float* __restrict__ out)
{
  __shared__ __align__(16) __bf16 wlds[2][DD * WPITCH];  // double-buffered Wh^T tile
  __shared__ float wh2s[NB];
  __shared__ float wh1s[ROWS_PER_WG];
  __shared__ float lrow[ROWS_PER_WG];

  const int bx  = blockIdx.x;
  const int bh  = bx >> 5;
  const int nb  = bx & 31;
  const int b   = bh >> 3;
  const int hd  = bh & 7;
  const int n0  = nb * ROWS_PER_WG;
  const int t   = threadIdx.x;
  const int lane = t & 31;
  const int w   = t >> 5;

  // stage Wh2 (all 4096) and Wh1 (this block's 128 rows)
  {
    const float* p = wh2 + bh * NB;
    for (int i = t * 4; i < NB; i += 1024)
      *(float4*)&wh2s[i] = *(const float4*)(p + i);
    if (t < ROWS_PER_WG) wh1s[t] = wh1[bh * NB + n0 + t];
  }

  // cooperative tile-0 load: thread -> (o = t>>2, 32B chunk)
  const __bf16* wtg = whT + (size_t)bh * DD * NB;
  const int lo = t >> 2;
  const int lc = (t & 3) * 16;
  v8bf r0, r1;
  {
    const __bf16* src = wtg + (size_t)lo * NB + lc;
    r0 = *(const v8bf*)(src);
    r1 = *(const v8bf*)(src + 8);
  }
  *(v8bf*)&wlds[0][lo * WPITCH + lc]     = r0;
  *(v8bf*)&wlds[0][lo * WPITCH + lc + 8] = r1;
  __syncthreads();

  const int half   = lane >> 4;
  const int lrow16 = lane & 15;
  const int nw0    = n0 + w * 16;
  const float m2v  = mono_f32(m2u[bh]);
  const float x1   = wh1s[w * 16 + lrow16];
  const float xm   = x1 + m2v;
  const float mn   = fmaxf(xm, LRELU_ALPHA * xm);  // exact row max of scores
  float lsum = 0.f;
  v8f acc[4] = {};

  for (int tt = 0; tt < NB / TM; ++tt) {
    const int cur = tt & 1;
    const int m0  = tt * TM;
    if (tt + 1 < NB / TM) {                       // prefetch next tile to regs
      const __bf16* src = wtg + (size_t)lo * NB + (m0 + TM) + lc;
      r0 = *(const v8bf*)(src);
      r1 = *(const v8bf*)(src + 8);
    }
#pragma unroll
    for (int s = 0; s < 2; ++s) {
      // A operand = P tile (exp of shifted leaky-relu scores), built in-lane
      v16bf av;
      const float* w2p = &wh2s[m0 + s * 32 + half * 8];
#pragma unroll
      for (int g = 0; g < 2; ++g) {
        const float4 f0 = *(const float4*)(w2p + g * 16);
        const float4 f1 = *(const float4*)(w2p + g * 16 + 4);
        float vv[8] = { f0.x, f0.y, f0.z, f0.w, f1.x, f1.y, f1.z, f1.w };
#pragma unroll
        for (int j = 0; j < 8; ++j) {
          const float x  = x1 + vv[j];
          const float sc = fmaxf(x, LRELU_ALPHA * x);
          const float p  = __expf(sc - mn);
          lsum += p;
          av[g * 8 + j] = (__bf16)p;
        }
      }
#pragma unroll
      for (int ct = 0; ct < 4; ++ct) {
        const __bf16* bp =
            &wlds[cur][(ct * 16 + lrow16) * WPITCH + s * 32 + half * 8];
        v8bf blo = *(const v8bf*)(bp);
        v8bf bhi = *(const v8bf*)(bp + 16);
        v16bf bv = __builtin_shufflevector(blo, bhi,
                     0,1,2,3,4,5,6,7,8,9,10,11,12,13,14,15);
        acc[ct] = __builtin_amdgcn_wmma_f32_16x16x32_bf16(
                    false, av, false, bv, (short)0, acc[ct], false, false);
      }
    }
    if (tt + 1 < NB / TM) {
      __bf16* dst = &wlds[1 - cur][lo * WPITCH + lc];
      *(v8bf*)(dst)     = r0;
      *(v8bf*)(dst + 8) = r1;
    }
    __syncthreads();
  }

  // softmax denominator: lanes L and L+16 hold complementary m-halves of row L&15
  const float ltot = lsum + __shfl_xor(lsum, 16);
  lrow[w * 16 + lrow16] = ltot;
  __syncthreads();

  float* outp = out + ((size_t)b * NB) * (HH * DD) + hd * DD;
#pragma unroll
  for (int r = 0; r < 8; ++r) {
    const int M = half * 8 + r;
    const float li = 1.0f / lrow[w * 16 + M];
    const int n = nw0 + M;
#pragma unroll
    for (int ct = 0; ct < 4; ++ct) {
      const float v = acc[ct][r] * li;
      const float e = v > 0.f ? v : (__expf(v) - 1.0f);   // ELU
      outp[(size_t)n * (HH * DD) + ct * 16 + lrow16] = e;
    }
  }
}

extern "C" void kernel_launch(void* const* d_in, const int* in_sizes, int n_in,
                              void* d_out, int out_size, void* d_ws, size_t ws_size,
                              hipStream_t stream) {
  (void)in_sizes; (void)n_in; (void)out_size; (void)ws_size;
  const float* h = (const float*)d_in[0];
  const float* W = (const float*)d_in[1];
  const float* a = (const float*)d_in[2];
  float* out = (float*)d_out;

  char* ws = (char*)d_ws;
  __bf16*   whT = (__bf16*)ws;                               // 16*64*4096*2 = 8 MiB
  float*    wh1 = (float*)(ws + 8u * 1024u * 1024u);         // 256 KiB
  float*    wh2 = (float*)(ws + 8u * 1024u * 1024u + 262144u);
  unsigned* m2u = (unsigned*)(ws + 8u * 1024u * 1024u + 2u * 262144u);

  k_init <<<1,   64,  0, stream>>>(m2u);
  k_pass1<<<512, 256, 0, stream>>>(h, W, a, whT, wh1, wh2, m2u);
  k_pass2<<<512, 256, 0, stream>>>(whT, wh1, wh2, m2u, out);
}
